// QLSTMPlus_65481071398345
// MI455X (gfx1250) — matrix operand order repaired and checked by reading, not compiled, not run on hardware
//
#include <hip/hip_runtime.h>

// ---- CDNA5 WMMA vector types ----
typedef __attribute__((ext_vector_type(16))) __bf16 v16bf;
typedef __attribute__((ext_vector_type(8)))  __bf16 v8bf;
typedef __attribute__((ext_vector_type(8)))  float  v8f;

#define T_STEPS  1024
#define BATCH    64
#define DIN      256
#define HID      256
#define KDIM     512          // D + H
#define NWG      16           // persistent workgroups (grid barrier members)
#define NTHREADS 256          // 8 wave32s
#define HSLICE   16           // H columns per workgroup
#define LDC      520          // combined LDS row stride (bf16), padded
#define LDW      520          // W-transpose LDS row stride (bf16), padded
#define LDG      68           // gates LDS row stride (f32), padded

__global__ void qlstm_init_ws(unsigned* ws) {
    if (threadIdx.x == 0) ws[0] = 0u;
}

// sigmoid via v_exp_f32 + v_rcp_f32 (no IEEE div sequence)
__device__ __forceinline__ float fast_sigmoid(float x) {
    return __builtin_amdgcn_rcpf(1.f + __expf(-x));
}
// tanh via exp/rcp, saturates correctly at +/-1 for large |x|
__device__ __forceinline__ float fast_tanh(float x) {
    float e = __expf(2.f * x);
    return 1.f - 2.f * __builtin_amdgcn_rcpf(e + 1.f);
}

__global__ __launch_bounds__(NTHREADS)
void qlstm_persistent_kernel(const float* __restrict__ x,      // [T,B,D]
                             const float* __restrict__ W,      // [512, 1024] row-major
                             const float* __restrict__ bias,   // [1024]
                             const float* __restrict__ beta_p, // scalar
                             const float* __restrict__ noise,  // [T,B,H]
                             float* __restrict__ out,          // outputs[T,B,H] ++ hx[B,H] ++ cx[B,H]
                             unsigned* __restrict__ bar)
{
    __shared__ __bf16 comb[BATCH * LDC];        // combined [64][512+pad] bf16
    __shared__ __bf16 Wt[4 * HSLICE * LDW];     // W slice transposed: [64 cols][512+pad] bf16
    __shared__ float  gatesL[BATCH * LDG];      // [64][64+pad] f32
    __shared__ float  blds[4 * HSLICE];

    const int tid  = threadIdx.x;
    const int wg   = blockIdx.x;       // 0..15 -> H cols [wg*16, wg*16+16)
    const int lane = tid & 31;
    const int wid  = tid >> 5;         // 0..7
    const int r16  = lane & 15;
    const int hh   = lane >> 4;        // lane half (0/1)

    const float beta = beta_p[0];

    // ---- one-time: load W slice into LDS, transposed, as bf16 ----
    // gate columns for local col c (0..63): (c/16)*HID + wg*16 + (c%16)
    for (int idx = tid; idx < 4 * HSLICE * KDIM; idx += NTHREADS) {
        int c   = idx / KDIM;
        int k   = idx % KDIM;
        int col = (c >> 4) * HID + wg * HSLICE + (c & 15);
        Wt[c * LDW + k] = (__bf16)W[k * (4 * HID) + col];
    }
    if (tid < 4 * HSLICE) {
        blds[tid] = bias[(tid >> 4) * HID + wg * HSLICE + (tid & 15)];
    }

    float creg[4]  = {0.f, 0.f, 0.f, 0.f};  // cell state, 4 elements per thread, all T steps
    float hlast[4] = {0.f, 0.f, 0.f, 0.f};

    for (int t = 0; t < T_STEPS; ++t) {
        // ---- stage combined = [x_t | h_{t-1}] into LDS as bf16 ----
        const float* xt    = x   + (size_t)t * (BATCH * DIN);
        const float* hprev = out + (size_t)(t - 1) * (BATCH * HID);  // used only when t>0
        #pragma unroll 4
        for (int j = 0; j < 32; ++j) {
            int c4  = tid + j * NTHREADS;   // float4 index over 64*512 floats
            int f   = c4 * 4;
            int row = f >> 9;               // /512
            int k   = f & 511;
            float4 v;
            if (k < DIN) {
                v = *(const float4*)(xt + row * DIN + k);
            } else if (t > 0) {
                v = *(const float4*)(hprev + row * HID + (k - DIN));
            } else {
                v = make_float4(0.f, 0.f, 0.f, 0.f);
            }
            union { __bf16 h[4]; uint2 u; } p;
            p.h[0] = (__bf16)v.x; p.h[1] = (__bf16)v.y;
            p.h[2] = (__bf16)v.z; p.h[3] = (__bf16)v.w;
            *(uint2*)&comb[row * LDC + k] = p.u;   // 8B aligned: LDC*2 and k*2 are multiples of 8
        }

        // ---- prefetch next step's streaming inputs (global_prefetch_b8) ----
        if (t + 1 < T_STEPS) {
            const float* xn = x + (size_t)(t + 1) * (BATCH * DIN);
            const float* nn = noise + (size_t)(t + 1) * (BATCH * HID);
            __builtin_prefetch(xn + tid * 64, 0, 1);  // 256B apart per lane
            __builtin_prefetch(nn + tid * 64, 0, 1);
        }
        __syncthreads();

        // ---- gate GEMM: [64 x 512] x [512 x 64] via v_wmma_f32_16x16x32_bf16 ----
        // 16 tiles (4 M-tiles x 4 gate-tiles), 2 per wave
        for (int s = 0; s < 2; ++s) {
            int tile = wid * 2 + s;
            int mi = tile >> 2;     // batch-row tile
            int ni = tile & 3;      // gate index (0=f,1=i,2=g,3=o)
            v8f acc = {};
            #pragma unroll 4
            for (int kc = 0; kc < KDIM / 32; ++kc) {
                // A fragment (16x32 bf16): lanes 0-15 K={kc*32+0..7, +16..23}, lanes 16-31 K={+8..15, +24..31}
                const __bf16* arow = &comb[(mi * 16 + r16) * LDC + kc * 32 + hh * 8];
                union { v16bf v; struct { v8bf lo, hi; } h2; } ua;
                ua.h2.lo = *(const v8bf*)(arow);
                ua.h2.hi = *(const v8bf*)(arow + 16);
                // B fragment (32x16 bf16): lane col = ni*16+r16, lanes 0-15 K={kc*32+0..7, +16..23}, lanes 16-31 K={+8..15, +24..31}
                const __bf16* brow = &Wt[(ni * 16 + r16) * LDW + kc * 32 + hh * 8];
                union { v16bf v; struct { v8bf lo, hi; } h2; } ub;
                ub.h2.lo = *(const v8bf*)(brow);
                ub.h2.hi = *(const v8bf*)(brow + 16);
                acc = __builtin_amdgcn_wmma_f32_16x16x32_bf16(
                          false, ua.v, false, ub.v, (short)0, acc, false, false);
            }
            // bias + branchless per-gate activation, scatter to gates LDS
            // gate g (ni==2) uses tanh(v) = 2*sigmoid(2v) - 1; f,i,o use sigmoid(v)
            float bb    = blds[ni * 16 + r16];
            float scale = (ni == 2) ? 2.f : 1.f;
            float mulv  = (ni == 2) ? 2.f : 1.f;
            float addv  = (ni == 2) ? -1.f : 0.f;
            #pragma unroll
            for (int j = 0; j < 8; ++j) {
                float val = acc[j] + bb;
                float sg  = fast_sigmoid(scale * val);
                float a   = __builtin_fmaf(sg, mulv, addv);
                int m = mi * 16 + j + 8 * hh;       // C layout: VGPR j -> M=j (+8 for upper lanes)
                gatesL[m * LDG + ni * 16 + r16] = a;
            }
        }
        __syncthreads();

        // ---- elementwise LSTM cell update (4 elements per thread) ----
        #pragma unroll
        for (int q = 0; q < 4; ++q) {
            int ei  = tid * 4 + q;      // over [64 rows][16 hcols]
            int row = ei >> 4;
            int hc  = ei & 15;
            float fg = gatesL[row * LDG +  0 + hc];
            float ig = gatesL[row * LDG + 16 + hc];
            float gg = gatesL[row * LDG + 32 + hc];
            float og = gatesL[row * LDG + 48 + hc];
            float cn = __builtin_fmaf(fg, creg[q], ig * gg);
            creg[q] = cn;
            size_t gi = (size_t)t * (BATCH * HID) + row * HID + wg * HSLICE + hc;
            float hn = __builtin_fmaf(og, fast_tanh(cn), beta * noise[gi]);
            hlast[q] = hn;
            out[gi]  = hn;              // outputs[t] doubles as h state for step t+1
        }

        // ---- grid barrier (monotonic counter in d_ws, per-launch init) ----
        __syncthreads();
        __threadfence();                // release h writes to device scope
        if (tid == 0) {
            atomicAdd(bar, 1u);
            unsigned target = (unsigned)(t + 1) * NWG;
            while (__hip_atomic_load(bar, __ATOMIC_RELAXED, __HIP_MEMORY_SCOPE_AGENT) < target) {
                __builtin_amdgcn_s_sleep(2);
            }
        }
        __syncthreads();
        __threadfence();                // acquire other WGs' h writes
    }

    // ---- final carry: hx, cx ----
    {
        const size_t base_h = (size_t)T_STEPS * BATCH * HID;
        const size_t base_c = base_h + (size_t)BATCH * HID;
        #pragma unroll
        for (int q = 0; q < 4; ++q) {
            int ei  = tid * 4 + q;
            int row = ei >> 4;
            int hc  = ei & 15;
            size_t o = (size_t)row * HID + wg * HSLICE + hc;
            out[base_h + o] = hlast[q];
            out[base_c + o] = creg[q];
        }
    }
}

extern "C" void kernel_launch(void* const* d_in, const int* in_sizes, int n_in,
                              void* d_out, int out_size, void* d_ws, size_t ws_size,
                              hipStream_t stream) {
    const float* x      = (const float*)d_in[0];   // inputs [T,B,D]
    const float* W      = (const float*)d_in[1];   // [512, 1024]
    const float* b      = (const float*)d_in[2];   // [1024]
    const float* beta   = (const float*)d_in[3];   // scalar
    const float* noise  = (const float*)d_in[4];   // [T,B,H]
    float*       out    = (float*)d_out;
    unsigned*    bar    = (unsigned*)d_ws;

    qlstm_init_ws<<<1, 32, 0, stream>>>(bar);
    qlstm_persistent_kernel<<<NWG, NTHREADS, 0, stream>>>(x, W, b, beta, noise, out, bar);
}
